// ParallelCoAttentionLayer_66511863546304
// MI455X (gfx1250) — matrix operand order repaired and checked by reading, not compiled, not run on hardware
//
#include <hip/hip_runtime.h>
#include <math.h>

// ParallelCoAttentionLayer for MI455X (gfx1250, wave32).
// One wave per node. All GEMMs via V_WMMA_F32_16X16X32_F16.
// Node-invariant operands pre-packed into exact WMMA lane layouts (prep pass);
// per-node p staged to LDS with GLOBAL_LOAD_ASYNC_TO_LDS_B128 and player_out
// written back with GLOBAL_STORE_ASYNC_FROM_LDS_B128 (fallback if builtins absent).

#define ASPECTS 16
#define ADIM    128
#define HID     2048
#define PROW    132    // padded row stride (floats) for staged p (bank-conflict free)
#define TSTR    17     // padded stride for 16-wide f32 tiles

typedef __attribute__((ext_vector_type(16))) _Float16 v16h;
typedef __attribute__((ext_vector_type(8)))  float    v8f;
typedef __attribute__((ext_vector_type(4)))  float    f32x4;

// pointer types expected by the gfx1250 async LDS builtins
typedef int i32x4v __attribute__((vector_size(16)));
typedef __attribute__((address_space(1))) i32x4v* gbl128_t;
typedef __attribute__((address_space(3))) i32x4v* lds128_t;

#if __has_builtin(__builtin_amdgcn_global_load_async_to_lds_b128) && \
    __has_builtin(__builtin_amdgcn_global_store_async_from_lds_b128)
#define USE_ASYNC 1
#else
#define USE_ASYNC 0
#endif

static __device__ __forceinline__ void wait_async0() {
#if __has_builtin(__builtin_amdgcn_s_wait_asynccnt)
  __builtin_amdgcn_s_wait_asynccnt(0);
#else
  asm volatile("s_wait_asynccnt 0x0" ::: "memory");
#endif
  asm volatile("" ::: "memory");
}
static __device__ __forceinline__ void wait_ds0() {
  asm volatile("" ::: "memory");
#if __has_builtin(__builtin_amdgcn_s_wait_dscnt)
  __builtin_amdgcn_s_wait_dscnt(0);
#else
  asm volatile("s_wait_dscnt 0x0" ::: "memory");
#endif
}

static __device__ __forceinline__ float fexp2(float x) {
#if __has_builtin(__builtin_amdgcn_exp2f)
  return __builtin_amdgcn_exp2f(x);
#else
  return exp2f(x);
#endif
}
static __device__ __forceinline__ float frcp(float x) {
#if __has_builtin(__builtin_amdgcn_rcpf)
  return __builtin_amdgcn_rcpf(x);
#else
  return 1.f / x;
#endif
}
// branch-free tanh: (e^{2x}-1)/(e^{2x}+1), clamped
static __device__ __forceinline__ float fast_tanh(float x) {
  const float cx = fminf(fmaxf(x, -10.f), 10.f);
  const float e  = fexp2(cx * 2.885390081777927f);  // 2*log2(e)
  return (e - 1.f) * frcp(e + 1.f);
}

static __device__ __forceinline__ v8f wmma_f16(v16h a, v16h b, v8f c) {
  return __builtin_amdgcn_wmma_f32_16x16x32_f16(false, a, false, b, (short)0, c,
                                                false, false);
}

// wave32 WMMA lane/K patterns
static __device__ __forceinline__ int akpat(int lane, int h) {
  return ((lane >> 4) << 3) + (h < 8 ? h : h + 8);
}
static __device__ __forceinline__ int bkpat(int lane, int h) {
  return ((lane >> 4) << 3) + (h < 8 ? h : h + 8);
}

// A fragment (16x32 f16): elem(m,k)=src[(m0+m)*rs + k*cs], zero if k>=klim
template <typename T>
static __device__ __forceinline__ v16h fragA(const T* src, int rs, int cs,
                                             int m0, int k0, int klim) {
  const int l = threadIdx.x & 31;
  const int m = m0 + (l & 15);
  v16h r;
#pragma unroll
  for (int h = 0; h < 16; ++h) {
    const int k = k0 + akpat(l, h);
    r[h] = (k < klim) ? (_Float16)src[m * rs + k * cs] : (_Float16)0.0f;
  }
  return r;
}
// B fragment (32x16 f16): elem(k,n)=src[k*rs + (n0+n)*cs], zero if k>=klim
template <typename T>
static __device__ __forceinline__ v16h fragB(const T* src, int rs, int cs,
                                             int k0, int n0, int klim) {
  const int l = threadIdx.x & 31;
  const int n = n0 + (l & 15);
  v16h r;
#pragma unroll
  for (int h = 0; h < 16; ++h) {
    const int k = k0 + bkpat(l, h);
    r[h] = (k < klim) ? (_Float16)src[k * rs + n * cs] : (_Float16)0.0f;
  }
  return r;
}
// C accumulator (16x16 f32) from strided buffer
template <typename T>
static __device__ __forceinline__ v8f loadC(const T* src, int rs, int cs,
                                            int m0, int n0) {
  const int l  = threadIdx.x & 31;
  const int n  = n0 + (l & 15);
  const int mb = m0 + (l >> 4) * 8;
  v8f c;
#pragma unroll
  for (int j = 0; j < 8; ++j) c[j] = (float)src[(mb + j) * rs + n * cs];
  return c;
}
static __device__ __forceinline__ void storeD(float* dst, int ld, int m0, int n0,
                                              v8f d) {
  const int l  = threadIdx.x & 31;
  const int n  = n0 + (l & 15);
  const int mb = m0 + (l >> 4) * 8;
#pragma unroll
  for (int j = 0; j < 8; ++j) dst[(mb + j) * ld + n] = d[j];
}

// packed fragment loads: [frag][lane][16 halfs] -> 32B contiguous per lane
static __device__ __forceinline__ v16h ldFrag(const _Float16* pk, int f) {
  const int lane = threadIdx.x & 31;
  return *(const v16h*)(pk + (((f << 5) + lane) << 4));
}
static __device__ __forceinline__ v8f ldFragC8(const float* pk, int f) {
  const int lane = threadIdx.x & 31;
  return *(const v8f*)(pk + (((f << 5) + lane) << 3));
}

// row softmax of a 16x16 tile (stride ld), over second index; branch-free exp
static __device__ __forceinline__ void softmax_rows16(float* S, int ld) {
  const int lane = threadIdx.x & 31;
  if (lane < 16) {
    float* row = S + lane * ld;
    float mx = row[0];
#pragma unroll
    for (int b = 1; b < 16; ++b) mx = fmaxf(mx, row[b]);
    float e[16];
    float s = 0.f;
#pragma unroll
    for (int b = 0; b < 16; ++b) {
      e[b] = fexp2((row[b] - mx) * 1.4426950408889634f);
      s += e[b];
    }
    const float inv = frcp(s);
#pragma unroll
    for (int b = 0; b < 16; ++b) row[b] = e[b] * inv;
  }
}

// ---------------- workspace layout (float offsets from ws) ----------------
// [0,2048)      teamAcc (f32, zeroed)
// [2048,4096)   wbT = w_b@t  [128][16] f32
// [4096,6144)   wtT = w_t@t  [128][16] f32
// [6144,8192)   wtC: 8 C-fragments of wtT  [8][32][8] f32
// halfs at ws+8192:
//   wpA  [32*512]  A-frags of w_p (tile t, kstep q; f = t*4+q)
//   wbB  [ 4*512]  B-frags of wbT
//   whtA [ 4*512]  A-frags of w_ht^T
//   whpA [ 4*512]  A-frags of w_hp^T
//   wtA  [ 8*512]  A-frags of wtT (K=16 zero-padded)
//   teamB[ 8*512]  B-frags of t^T (K=16 zero-padded)

__global__ void k_prep1(const float* __restrict__ team,
                        const float* __restrict__ w_b,
                        const float* __restrict__ w_t,
                        float* __restrict__ ws) {
  const int tid = blockIdx.x * blockDim.x + threadIdx.x;
  if (tid >= 3 * HID) return;
  if (tid < HID) { ws[tid] = 0.f; return; }
  const int o   = tid - HID;
  const int idx = o & (HID - 1);   // dd*16 + a
  const int dd  = idx >> 4;
  const int a   = idx & 15;
  const float* W = (o < HID) ? w_b : w_t;
  float s = 0.f;
  for (int e = 0; e < ADIM; ++e) s += W[dd * ADIM + e] * team[a * ADIM + e];
  ws[HID + o] = s;
}

__global__ void k_pack(const float* __restrict__ team,
                       const float* __restrict__ w_p,
                       const float* __restrict__ w_ht,
                       const float* __restrict__ w_hp,
                       float* __restrict__ ws) {
  const float* wbT = ws + 2048;
  const float* wtT = ws + 4096;
  float* wtC = ws + 6144;
  _Float16* hp    = (_Float16*)(ws + 8192);
  _Float16* wpA   = hp;
  _Float16* wbB   = hp + 16384;
  _Float16* whtA  = hp + 18432;
  _Float16* whpA  = hp + 20480;
  _Float16* wtA   = hp + 22528;
  _Float16* teamB = hp + 26624;

  const int tid = blockIdx.x * blockDim.x + threadIdx.x;
  if (tid < 16384) {                       // wpA
    const int f = tid >> 9, r = tid & 511, lane = r >> 4, h = r & 15;
    const int t = f >> 2, q = f & 3;
    const int m = t * 16 + (lane & 15);
    const int k = q * 32 + akpat(lane, h);
    wpA[tid] = (_Float16)w_p[m * ADIM + k];
  } else if (tid < 18432) {                // wbB
    const int o = tid - 16384;
    const int q = o >> 9, r = o & 511, lane = r >> 4, h = r & 15;
    const int n = lane & 15, k = q * 32 + bkpat(lane, h);
    wbB[o] = (_Float16)wbT[k * 16 + n];
  } else if (tid < 20480) {                // whtA (w_ht^T)
    const int o = tid - 18432;
    const int q = o >> 9, r = o & 511, lane = r >> 4, h = r & 15;
    const int m = lane & 15, k = q * 32 + akpat(lane, h);
    whtA[o] = (_Float16)w_ht[k * 16 + m];
  } else if (tid < 22528) {                // whpA (w_hp^T)
    const int o = tid - 20480;
    const int q = o >> 9, r = o & 511, lane = r >> 4, h = r & 15;
    const int m = lane & 15, k = q * 32 + akpat(lane, h);
    whpA[o] = (_Float16)w_hp[k * 16 + m];
  } else if (tid < 26624) {                // wtA (K=16 padded)
    const int o = tid - 22528;
    const int t = o >> 9, r = o & 511, lane = r >> 4, h = r & 15;
    const int m = t * 16 + (lane & 15), k = akpat(lane, h);
    wtA[o] = (k < 16) ? (_Float16)wtT[m * 16 + k] : (_Float16)0.f;
  } else if (tid < 30720) {                // teamB (t^T, K=16 padded)
    const int o = tid - 26624;
    const int t = o >> 9, r = o & 511, lane = r >> 4, h = r & 15;
    const int n = t * 16 + (lane & 15), k = bkpat(lane, h);
    teamB[o] = (k < 16) ? (_Float16)team[k * ADIM + n] : (_Float16)0.f;
  } else if (tid < 32768) {                // wtC (f32 C-fragments)
    const int o = tid - 30720;
    const int t = o >> 8, r = o & 255, lane = r >> 3, j = r & 7;
    const int n = lane & 15, m = t * 16 + ((lane >> 4) << 3) + j;
    wtC[o] = wtT[m * 16 + n];
  }
}

__global__ void __launch_bounds__(64)
k_coattn(const float* __restrict__ player,
         const _Float16* __restrict__ wpA,  const _Float16* __restrict__ wbB,
         const _Float16* __restrict__ whtA, const _Float16* __restrict__ whpA,
         const _Float16* __restrict__ wtA,  const _Float16* __restrict__ teamB,
         const float* __restrict__ wtC,
         float* __restrict__ teamAcc, float* __restrict__ outP, int n_nodes) {
  __shared__ __align__(16) float sP [2][16 * PROW];   // p row-major, padded
  __shared__ __align__(16) float sWp[2][128 * TSTR];  // w_p @ p^T
  __shared__ __align__(16) float sCo[2][128 * 16];    // co / out staging
  __shared__ __align__(16) float sC [2][16 * TSTR];   // C
  __shared__ __align__(16) float sS [2][16 * TSTR];   // scores / attention
  __shared__ float sAcc[HID];

  const int w    = threadIdx.x >> 5;
  const int lane = threadIdx.x & 31;
  float* P  = sP[w];
  float* Wp = sWp[w];
  float* Co = sCo[w];
  float* Cm = sC[w];
  float* Sm = sS[w];

  for (int i = threadIdx.x; i < HID; i += 64) sAcc[i] = 0.f;
  __syncthreads();

  const int gw    = (blockIdx.x << 1) + w;
  const int nwave = gridDim.x << 1;
  const int iters = (n_nodes + nwave - 1) / nwave;

  for (int it = 0; it < iters; ++it) {
    const int node = gw + it * nwave;
    if (node >= n_nodes) continue;  // wave-uniform

    const float* pn = player + (size_t)node * HID;

    // ---- stage p [16][128] -> LDS padded [16][PROW] ----
#if USE_ASYNC
#pragma unroll
    for (int i = 0; i < 16; ++i) {
      const int g = lane + (i << 5);  // float4 index
      __builtin_amdgcn_global_load_async_to_lds_b128(
          (gbl128_t)(pn + (g << 2)),
          (lds128_t)(P + (g >> 5) * PROW + ((g & 31) << 2)), 0, 0);
    }
    wait_async0();
#else
#pragma unroll
    for (int i = 0; i < 16; ++i) {
      const int g = lane + (i << 5);
      const f32x4 v = ((const f32x4*)pn)[g];
      *(f32x4*)(P + (g >> 5) * PROW + ((g & 31) << 2)) = v;
    }
#endif

    // p fragments (A for C-GEMM; B for Wp-GEMM), hoisted across tiles
    v16h pA[4], pB[4];
#pragma unroll
    for (int q = 0; q < 4; ++q) {
      pA[q] = fragA(P, PROW, 1, 0, q * 32, ADIM);  // elem(a,d)=p[a][d]
      pB[q] = fragB(P, 1, PROW, q * 32, 0, ADIM);  // elem(d,a)=p[a][d]
    }

    // ---- C = tanh(p @ wb_t) [16x16] ----
    {
      v8f c = {};
#pragma unroll
      for (int q = 0; q < 4; ++q) c = wmma_f16(pA[q], ldFrag(wbB, q), c);
#pragma unroll
      for (int j = 0; j < 8; ++j) c[j] = fast_tanh(c[j]);
      storeD(Cm, TSTR, 0, 0, c);
    }

    // ---- Wp = w_p @ p^T [128x16] ----
#pragma unroll
    for (int t = 0; t < 8; ++t) {
      v8f c = {};
#pragma unroll
      for (int q = 0; q < 4; ++q) c = wmma_f16(ldFrag(wpA, (t << 2) + q), pB[q], c);
      storeD(Wp, TSTR, t << 4, 0, c);
    }

    // ================= team path =================
    {
      const v16h bC = fragB(Cm, TSTR, 1, 0, 0, 16);
#pragma unroll
      for (int t = 0; t < 8; ++t) {
        v8f c = ldFragC8(wtC, t);
        c = wmma_f16(fragA(Wp, TSTR, 1, t << 4, 0, 16), bC, c);
#pragma unroll
        for (int j = 0; j < 8; ++j) c[j] = fast_tanh(c[j]);
        storeD(Co, 16, t << 4, 0, c);
      }
    }
    {
      v8f c = {};
#pragma unroll
      for (int q = 0; q < 4; ++q)
        c = wmma_f16(ldFrag(whtA, q), fragB(Co, 16, 1, q * 32, 0, 128), c);
      storeD(Sm, TSTR, 0, 0, c);
    }
    softmax_rows16(Sm, TSTR);
    {
      const v16h aS = fragA(Sm, TSTR, 1, 0, 0, 16);
      const int mb = (lane >> 4) << 3, nn = lane & 15;
#pragma unroll
      for (int t = 0; t < 8; ++t) {
        v8f c = {};
        c = wmma_f16(aS, ldFrag(teamB, t), c);
#pragma unroll
        for (int j = 0; j < 8; ++j)
          atomicAdd(&sAcc[(mb + j) * ADIM + (t << 4) + nn], c[j]);
      }
    }

    // ================= player path =================
    {
      const v16h bCt = fragB(Cm, 1, TSTR, 0, 0, 16);  // C^T
#pragma unroll
      for (int t = 0; t < 8; ++t) {
        v8f c = loadC(Wp, TSTR, 1, t << 4, 0);
        c = wmma_f16(ldFrag(wtA, t), bCt, c);
#pragma unroll
        for (int j = 0; j < 8; ++j) c[j] = fast_tanh(c[j]);
        storeD(Co, 16, t << 4, 0, c);
      }
    }
    {
      v8f c = {};
#pragma unroll
      for (int q = 0; q < 4; ++q)
        c = wmma_f16(ldFrag(whpA, q), fragB(Co, 16, 1, q * 32, 0, 128), c);
      storeD(Sm, TSTR, 0, 0, c);
    }
    softmax_rows16(Sm, TSTR);
    {
      const v16h aS = fragA(Sm, TSTR, 1, 0, 0, 16);
      float* OutB = Co;  // reuse as [16][128] staging
#pragma unroll
      for (int t = 0; t < 8; ++t) {
        v8f c = {};
        c = wmma_f16(aS, fragB(P, PROW, 1, 0, t << 4, 16), c);
        storeD(OutB, ADIM, 0, t << 4, c);
      }
      float* po = outP + (size_t)node * HID;
#if USE_ASYNC
      wait_ds0();
#pragma unroll
      for (int i = 0; i < 16; ++i) {
        const int g = lane + (i << 5);
        __builtin_amdgcn_global_store_async_from_lds_b128(
            (gbl128_t)(po + (g << 2)), (lds128_t)(OutB + (g << 2)), 0, 0);
      }
      wait_async0();
#else
#pragma unroll
      for (int i = 0; i < 16; ++i) {
        const int g = lane + (i << 5);
        ((f32x4*)po)[g] = *(const f32x4*)(OutB + (g << 2));
      }
#endif
    }
  }

  __syncthreads();
  for (int i = threadIdx.x; i < HID; i += 64) atomicAdd(&teamAcc[i], sAcc[i]);
}

__global__ void k_finalize(const float* __restrict__ acc, float* __restrict__ out,
                           float inv_n) {
  const int i = blockIdx.x * blockDim.x + threadIdx.x;
  if (i < HID) out[i] = acc[i] * inv_n;
}

extern "C" void kernel_launch(void* const* d_in, const int* in_sizes, int n_in,
                              void* d_out, int out_size, void* d_ws, size_t ws_size,
                              hipStream_t stream) {
  const float* team   = (const float*)d_in[0];
  const float* player = (const float*)d_in[1];
  const float* w_b    = (const float*)d_in[2];
  const float* w_t    = (const float*)d_in[3];
  const float* w_p    = (const float*)d_in[4];
  const float* w_ht   = (const float*)d_in[5];
  const float* w_hp   = (const float*)d_in[6];
  float* out = (float*)d_out;
  float* ws  = (float*)d_ws;   // needs ~94 KB

  const int n = in_sizes[1] / HID;

  float*    teamAcc = ws;
  float*    wtC     = ws + 6144;
  _Float16* hp      = (_Float16*)(ws + 8192);
  _Float16* wpA     = hp;
  _Float16* wbB     = hp + 16384;
  _Float16* whtA    = hp + 18432;
  _Float16* whpA    = hp + 20480;
  _Float16* wtA     = hp + 22528;
  _Float16* teamB   = hp + 26624;

  k_prep1<<<(3 * HID + 255) / 256, 256, 0, stream>>>(team, w_b, w_t, ws);
  k_pack<<<128, 256, 0, stream>>>(team, w_p, w_ht, w_hp, ws);
  k_coattn<<<1024, 64, 0, stream>>>(player, wpA, wbB, whtA, whpA, wtA, teamB,
                                    wtC, teamAcc, out + HID, n);
  k_finalize<<<(HID + 255) / 256, 256, 0, stream>>>(teamAcc, out,
                                                    1.0f / (float)n);
}
